// PixelwiseContrastiveLoss_32358283608354
// MI455X (gfx1250) — compile-verified, matrix operand order checked by reading, not run on hardware
//
#include <hip/hip_runtime.h>
#include <hip/hip_bf16.h>
#include <math.h>

typedef __attribute__((ext_vector_type(16))) _Float16 v16h;
typedef __attribute__((ext_vector_type(8)))  _Float16 v8h;
typedef __attribute__((ext_vector_type(8)))  float    v8f;

#define B_    8
#define C_    128
#define H_    256
#define W_    256
#define NPOS  2048
#define NNEG  8192
#define NTOT  (NPOS + NNEG)
#define EPSF  1e-6f

#define NTILE_M   (NPOS / 16)          // 128
#define NTILE_N   ((NPOS + NNEG) / 16) // 640 (first 128 are pos cols, rest neg)
#define NCHUNK    16                   // n-tiles per block sweep
#define NGROUPS   (NTILE_N / NCHUNK)   // 40
#define MWAVES    8                    // m-tiles (waves) per block
#define NTPB      2                    // n-tiles per staged buffer
#define NSTAGES   (NCHUNK / NTPB)      // 8

// LDS B rows: 128 f16 (256B) padded to 272B -> conflict-free half-wave reads
#define TROW      272
#define BUFROWS   (16 * NTPB)          // 32 rows per buffer
#define BUFBYTES  (BUFROWS * TROW)     // 8704 B per buffer

// generic pointer -> raw LDS byte offset (addrspace(3) ptrtoint)
__device__ __forceinline__ unsigned lds_off(const void* p) {
  return (unsigned)(unsigned long long)
      (const __attribute__((address_space(3))) char*)p;
}

// ---------------------------------------------------------------------------
// Kernel 0: zero the row-sum accumulators (rs_pos[2048] ++ rs_neg[2048])
// ---------------------------------------------------------------------------
__global__ void pcl_zero(float* __restrict__ p, int n) {
  int i = blockIdx.x * blockDim.x + threadIdx.x;
  if (i < n) p[i] = 0.0f;
}

// ---------------------------------------------------------------------------
// Kernel 1: gather pixels + L2 normalize -> f16 matrices [vec][C] row-major
// ---------------------------------------------------------------------------
__global__ __launch_bounds__(128)
void pcl_gather_normalize(const float* __restrict__ map,
                          const int* __restrict__ pb, const int* __restrict__ ph,
                          const int* __restrict__ pw,
                          const int* __restrict__ nb, const int* __restrict__ nh,
                          const int* __restrict__ nw,
                          _Float16* __restrict__ pos_n,
                          _Float16* __restrict__ neg_n) {
  __shared__ float red[128];
  const int vec = blockIdx.x;
  const int c   = threadIdx.x;

  int b, h, w;
  _Float16* out;
  if (vec < NPOS) {
    b = pb[vec]; h = ph[vec]; w = pw[vec];
    out = pos_n + (size_t)vec * C_;
  } else {
    const int v = vec - NPOS;
    b = nb[v]; h = nh[v]; w = nw[v];
    out = neg_n + (size_t)v * C_;
  }

  const float x = map[(((size_t)b * C_ + c) * H_ + h) * W_ + w];
  red[c] = x * x;
  __syncthreads();
  #pragma unroll
  for (int s = 64; s > 0; s >>= 1) {
    if (c < s) red[c] += red[c + s];
    __syncthreads();
  }
  const float inv = 1.0f / fmaxf(sqrtf(red[0]), EPSF);
  out[c] = (_Float16)(x * inv);
}

// ---------------------------------------------------------------------------
// Kernel 2: fused WMMA GEMM + exp + row-sum.
// Block = 8 waves; wave w owns m-tile (mBlock*8+w). All waves share 32-col
// B buffers staged in LDS via async load-to-LDS (double buffered).
// ---------------------------------------------------------------------------
__global__ __launch_bounds__(256)
void pcl_sim_expsum(const _Float16* __restrict__ pos_n,
                    const _Float16* __restrict__ neg_n,
                    float* __restrict__ rs_pos,
                    float* __restrict__ rs_neg) {
  __shared__ __align__(16) char smem[2 * BUFBYTES];

  const int wave = threadIdx.x >> 5;
  const int lane = threadIdx.x & 31;
  const int mBlock = blockIdx.x / NGROUPS;      // 0..15
  const int group  = blockIdx.x % NGROUPS;      // 0..39
  const int nTileBase = group * NCHUNK;

  const bool isPos = (nTileBase < NPOS / 16);
  const _Float16* __restrict__ Bm = isPos ? pos_n : neg_n;
  const int colBase = (isPos ? nTileBase : nTileBase - NPOS / 16) * 16;
  float* __restrict__ rs = isPos ? rs_pos : rs_neg;

  const int laneRow = lane & 15;            // row of A / col of B in tile
  const int laneK   = (lane >> 4) * 16;     // K sub-block for this half-wave
  const int mBase   = (mBlock * MWAVES + wave) * 16;

  // A fragments (16x128 f16) once per wave; contiguous 32B per lane
  v16h a[4];
  const _Float16* Aptr = pos_n + (size_t)(mBase + laneRow) * C_ + laneK;
  #pragma unroll
  for (int kb = 0; kb < 4; ++kb)
    a[kb] = *(const v16h*)(Aptr + kb * 32);

  // async staging: per stage, thread t copies 2x16B (rows sr and sr+16)
  const unsigned smemBase = lds_off(smem);
  const int sr = threadIdx.x >> 4;          // 0..15
  const int ss = threadIdx.x & 15;          // 0..15

  const char* g0 = (const char*)(Bm + (size_t)(colBase + sr) * C_) + ss * 16;
  const unsigned l0 = smemBase + sr * TROW + ss * 16;

  #define STAGE(st)                                                          \
    do {                                                                     \
      const char* ga = g0 + (size_t)(st) * (BUFROWS * C_ * 2);               \
      const char* gb = ga + 16 * C_ * 2;                                     \
      unsigned la = l0 + ((st) & 1) * BUFBYTES;                              \
      unsigned lb = la + 16 * TROW;                                          \
      asm volatile("global_load_async_to_lds_b128 %0, %1, off"               \
                   ::"v"(la), "v"(ga) : "memory");                           \
      asm volatile("global_load_async_to_lds_b128 %0, %1, off"               \
                   ::"v"(lb), "v"(gb) : "memory");                           \
    } while (0)

  STAGE(0);

  v8f acc = {};   // per-lane running exp-sums
  const char* sBase = smem + laneRow * TROW + laneK * 2;

  for (int st = 0; st < NSTAGES; ++st) {
    if (st + 1 < NSTAGES) {
      STAGE(st + 1);
      asm volatile("s_wait_asynccnt 0x2" ::: "memory"); // buffer st complete
    } else {
      asm volatile("s_wait_asynccnt 0x0" ::: "memory");
    }
    __syncthreads();   // all waves' portions of buffer st visible

    const char* sb0 = sBase + (st & 1) * BUFBYTES;
    #pragma unroll
    for (int t2 = 0; t2 < NTPB; ++t2) {
      const char* sb = sb0 + t2 * (16 * TROW);
      // hoist all 8 fragment loads so they clause and overlap the WMMAs
      v8h lo[4], hi[4];
      #pragma unroll
      for (int kb = 0; kb < 4; ++kb) {
        lo[kb] = *(const v8h*)(sb + kb * 64);
        hi[kb] = *(const v8h*)(sb + kb * 64 + 16);
      }
      v8f cfrag = {};
      #pragma unroll
      for (int kb = 0; kb < 4; ++kb) {
        v16h bf = __builtin_shufflevector(lo[kb], hi[kb], 0, 1, 2, 3, 4, 5, 6,
                                          7, 8, 9, 10, 11, 12, 13, 14, 15);
        cfrag = __builtin_amdgcn_wmma_f32_16x16x32_f16(
            /*neg_a=*/false, a[kb], /*neg_b=*/false, bf,
            /*c_mod=*/(short)0, cfrag, /*reuse_a=*/false, /*reuse_b=*/false);
      }
      #pragma unroll
      for (int r = 0; r < 8; ++r) acc[r] += __expf(cfrag[r]);
    }

    __syncthreads();   // all waves done reading buffer st before reuse
  }

  // Butterfly-reduce each element across the 16 lanes holding one row's N vals
  #pragma unroll
  for (int r = 0; r < 8; ++r) {
    float v = acc[r];
    #pragma unroll
    for (int off = 1; off < 16; off <<= 1) v += __shfl_xor(v, off, 32);
    acc[r] = v;
  }
  if ((lane & 15) == 0) {
    const int rowOff = (lane >> 4) * 8;     // lane0 -> M=r, lane16 -> M=8+r
    #pragma unroll
    for (int r = 0; r < 8; ++r)
      atomicAdd(&rs[mBase + rowOff + r], acc[r]);
  }
}

// ---------------------------------------------------------------------------
// Kernel 3: nll = -mean(log( (posSum - e) / (posSum - e + negSum) ))
// ---------------------------------------------------------------------------
__global__ __launch_bounds__(256)
void pcl_finalize(const float* __restrict__ rs_pos,
                  const float* __restrict__ rs_neg,
                  float* __restrict__ out) {
  __shared__ float red[256];
  const int t = threadIdx.x;
  const float expOne = 2.718281828459045f;
  float s = 0.0f;
  for (int i = t; i < NPOS; i += 256) {
    const float p   = rs_pos[i] - expOne;
    const float n   = rs_neg[i];
    const float lik = p / (p + n);
    s += __logf(lik);
  }
  red[t] = s;
  __syncthreads();
  #pragma unroll
  for (int st = 128; st > 0; st >>= 1) {
    if (t < st) red[t] += red[t + st];
    __syncthreads();
  }
  if (t == 0) out[0] = -red[0] / (float)NPOS;
}

// ---------------------------------------------------------------------------
extern "C" void kernel_launch(void* const* d_in, const int* in_sizes, int n_in,
                              void* d_out, int out_size, void* d_ws, size_t ws_size,
                              hipStream_t stream) {
  const float* map = (const float*)d_in[0];
  const int* pb = (const int*)d_in[1];
  const int* ph = (const int*)d_in[2];
  const int* pw = (const int*)d_in[3];
  const int* nb = (const int*)d_in[4];
  const int* nh = (const int*)d_in[5];
  const int* nw = (const int*)d_in[6];
  float* out = (float*)d_out;

  // Workspace layout
  char* ws = (char*)d_ws;
  _Float16* pos_n = (_Float16*)ws;                           // 2048*128 f16
  _Float16* neg_n = pos_n + (size_t)NPOS * C_;               // 8192*128 f16
  float* rs_pos = (float*)(ws + (size_t)NTOT * C_ * sizeof(_Float16));
  float* rs_neg = rs_pos + NPOS;                             // contiguous

  // 0) zero both row-sum accumulators (2*2048 floats, contiguous)
  pcl_zero<<<(2 * NPOS + 255) / 256, 256, 0, stream>>>(rs_pos, 2 * NPOS);

  // 1) gather + normalize -> f16
  pcl_gather_normalize<<<NTOT, 128, 0, stream>>>(map, pb, ph, pw, nb, nh, nw,
                                                 pos_n, neg_n);

  // 2) fused WMMA similarity + exp row sums
  //    blocks = (NTILE_M/8) * NGROUPS = 16*40 = 640, each 8 waves
  pcl_sim_expsum<<<(NTILE_M / MWAVES) * NGROUPS, 256, 0, stream>>>(
      pos_n, neg_n, rs_pos, rs_neg);

  // 3) finalize NLL
  pcl_finalize<<<1, 256, 0, stream>>>(rs_pos, rs_neg, out);
}
